// PhaseSimilarityTransformerLayer_26542897889659
// MI455X (gfx1250) — matrix operand order repaired and checked by reading, not compiled, not run on hardware
//
#include <hip/hip_runtime.h>
#include <hip/hip_bf16.h>
#include <math.h>

typedef __bf16 bf16;
typedef __bf16 bf16x8 __attribute__((ext_vector_type(8)));
typedef __bf16 v16bf  __attribute__((ext_vector_type(16)));
typedef float  v8f    __attribute__((ext_vector_type(8)));
typedef float  v4f    __attribute__((ext_vector_type(4)));
typedef float  v2f    __attribute__((ext_vector_type(2)));
typedef unsigned int u32x4 __attribute__((ext_vector_type(4)));
typedef int    i32x4  __attribute__((ext_vector_type(4)));
typedef int    i32x8  __attribute__((ext_vector_type(8)));

#define D_MODEL 1024
#define NHEAD   16
#define HEAD_DIM 64
#define PHASE_DIM 64
#define BB      4
#define SEQ     1024
#define MTOT    (BB * SEQ)   // 4096 flattened rows for projections

static __device__ __forceinline__ v16bf cat8(bf16x8 lo, bf16x8 hi) {
  return __builtin_shufflevector(lo, hi, 0,1,2,3,4,5,6,7,8,9,10,11,12,13,14,15);
}

// ---- Tensor Data Mover: stage a tile_h x tile_w (bf16) tile into LDS ----
// D# built per CDNA5 ISA sect.8. Tracked with TENSORcnt.
static __device__ __forceinline__ void tdm_load_tile_2d(
    const bf16* gptr, bf16* lptr, unsigned tile_w, unsigned tile_h,
    unsigned long long row_stride_elems) {
  unsigned long long ga = (unsigned long long)(uintptr_t)gptr;
  unsigned lds_off = (unsigned)(uintptr_t)lptr;   // LDS (AS3) byte offset
  const unsigned td0 = 0x40000000u;               // huge tensor dims: no OOB clip
  const unsigned td1 = 0x40000000u;
  u32x4 g0;
  g0[0] = 1u;                                     // count=1 (valid), user mode
  g0[1] = lds_off;                                // lds_addr [63:32]
  g0[2] = (unsigned)ga;                           // global_addr [95:64]
  g0[3] = (unsigned)((ga >> 32) & 0x01FFFFFFu) | (2u << 30); // addr[56:32], type=2
  i32x8 g1;
  g1[0] = (int)(1u << 16);                        // wg_mask=0, data_size=1 (2B)
  g1[1] = (int)((td0 & 0xFFFFu) << 16);           // tensor_dim0[15:0]
  g1[2] = (int)((td0 >> 16) | ((td1 & 0xFFFFu) << 16));   // dim0[31:16], dim1[15:0]
  g1[3] = (int)((td1 >> 16) | (tile_w << 16));    // dim1[31:16], tile_dim0
  g1[4] = (int)tile_h;                            // tile_dim1, tile_dim2=0
  g1[5] = (int)(unsigned)(row_stride_elems & 0xFFFFFFFFull); // dim0_stride lo
  g1[6] = (int)(unsigned)((row_stride_elems >> 32) & 0xFFFFull); // stride hi, dim1_stride=0
  g1[7] = 0;
  i32x4 gz = {0, 0, 0, 0};
#if defined(__clang_major__) && (__clang_major__ >= 23)
  i32x8 gz8 = {0, 0, 0, 0, 0, 0, 0, 0};
  __builtin_amdgcn_tensor_load_to_lds(g0, g1, gz, gz, gz8, 0);
#else
  __builtin_amdgcn_tensor_load_to_lds(g0, g1, gz, gz, 0);
#endif
}

// ---------------- f32 -> bf16 pack ----------------
__global__ __launch_bounds__(256) void cvt_f32_bf16_kernel(
    const float* __restrict__ in, bf16* __restrict__ out, int n) {
  int i = blockIdx.x * blockDim.x + threadIdx.x;
  if (i < n) out[i] = (bf16)in[i];
}

// ---------------- phase L2 normalize (one wave per row of 64) ----------------
__global__ __launch_bounds__(32) void phase_norm_kernel(
    const float* __restrict__ phase, float* __restrict__ pn) {
  const int row  = blockIdx.x;
  const int lane = threadIdx.x;
  const float* p = phase + (size_t)row * PHASE_DIM;
  float a = p[lane];
  float b = p[lane + 32];
  float s = a * a + b * b;
#pragma unroll
  for (int off = 16; off > 0; off >>= 1) s += __shfl_xor(s, off, 32);
  const float inv = 1.0f / fmaxf(sqrtf(s), 1e-12f);
  float* o = pn + (size_t)row * PHASE_DIM;
  o[lane]      = a * inv;
  o[lane + 32] = b * inv;
}

// ---- NT GEMM with TDM double-buffered LDS staging -------------------------
// out[m,n] = (sum_k A[m,k]*W[n,k] + bias[n]) * scale
// Block = 4 waves = 64x64 output tile; K stepped by 32 with TDM prefetch.
template <bool BF16OUT>
__global__ __launch_bounds__(128) void gemm_nt_tdm_kernel(
    const bf16* __restrict__ A, const bf16* __restrict__ W,
    const float* __restrict__ bias, void* __restrict__ outp, float scale) {
  __shared__ bf16 sA[2][64 * 32];
  __shared__ bf16 sB[2][64 * 32];

  const int lane = threadIdx.x & 31;
  const int wv   = threadIdx.x >> 5;
  const int lm   = lane & 15;
  const int hi   = lane >> 4;
  const int m0   = blockIdx.y * 64;
  const int n0   = blockIdx.x * 64;

  const bf16* aTile = A + (size_t)m0 * D_MODEL;
  const bf16* bTile = W + (size_t)n0 * D_MODEL;

  if (threadIdx.x < 32) {                 // wave 0 drives the TDM
    tdm_load_tile_2d(aTile, sA[0], 32, 64, D_MODEL);
    tdm_load_tile_2d(bTile, sB[0], 32, 64, D_MODEL);
  }

  v8f acc[4] = {};
#pragma unroll 1
  for (int kk0 = 0; kk0 < D_MODEL; kk0 += 32) {
    const int buf = (kk0 >> 5) & 1;
    if (threadIdx.x < 32) {
      if (kk0 + 32 < D_MODEL) {
        tdm_load_tile_2d(aTile + kk0 + 32, sA[buf ^ 1], 32, 64, D_MODEL);
        tdm_load_tile_2d(bTile + kk0 + 32, sB[buf ^ 1], 32, 64, D_MODEL);
        __builtin_amdgcn_s_wait_tensorcnt(2);   // current pair landed
      } else {
        __builtin_amdgcn_s_wait_tensorcnt(0);
      }
    }
    __syncthreads();

    const bf16* arow = &sA[buf][(wv * 16 + lm) * 32 + hi * 8];
    v16bf af = cat8(*(const bf16x8*)(arow), *(const bf16x8*)(arow + 16));
#pragma unroll
    for (int f = 0; f < 4; ++f) {
      const bf16* brow = &sB[buf][(f * 16 + lm) * 32 + hi * 8];
      v16bf bfg = cat8(*(const bf16x8*)(brow), *(const bf16x8*)(brow + 16));
      acc[f] = __builtin_amdgcn_wmma_f32_16x16x32_bf16(
          false, af, false, bfg, (short)0, acc[f], false, false);
    }
    __syncthreads();                       // done reading buf before reuse
  }

#pragma unroll
  for (int f = 0; f < 4; ++f) {
    const int col = n0 + f * 16 + lm;
    const float bc = bias[col];
#pragma unroll
    for (int r = 0; r < 8; ++r) {
      const int row = m0 + wv * 16 + r + hi * 8;
      const float v = (acc[f][r] + bc) * scale;
      if (BF16OUT) ((bf16*)outp)[(size_t)row * D_MODEL + col] = (bf16)v;
      else         ((float*)outp)[(size_t)row * D_MODEL + col] = v;
    }
  }
}

// ---------------- V transpose: Vt[b][h][d][k] = V[b][k][h*64+d] ----------------
__global__ __launch_bounds__(256) void transpose_v_kernel(
    const bf16* __restrict__ V, bf16* __restrict__ Vt) {
  const int i = blockIdx.x * blockDim.x + threadIdx.x;   // 0 .. 4M-1
  const int k = i & (SEQ - 1);
  const int d = (i >> 10) & (HEAD_DIM - 1);
  const int h = (i >> 16) & (NHEAD - 1);
  const int b = i >> 20;
  Vt[i] = V[((size_t)(b * SEQ + k)) * D_MODEL + h * HEAD_DIM + d];
}

// ---------------- logits: phase-bias (f32 WMMA) + per-head QK^T (bf16 WMMA) ----
__global__ __launch_bounds__(32) void logits_kernel(
    const bf16* __restrict__ Q, const bf16* __restrict__ Kp,
    const float* __restrict__ pn, const float* __restrict__ ps_ptr,
    float* __restrict__ attnW) {
  const int lane = threadIdx.x;
  const int lm = lane & 15;
  const int hi = lane >> 4;
  const int k0 = blockIdx.x * 16;
  const int q0 = blockIdx.y * 16;
  const int b  = blockIdx.z;
  const float ps = ps_ptr[0];

  const float* pa = pn + ((size_t)b * SEQ + q0 + lm) * PHASE_DIM;
  const float* pb = pn + ((size_t)b * SEQ + k0 + lm) * PHASE_DIM;
  v8f bias = {};
#pragma unroll
  for (int ks = 0; ks < PHASE_DIM; ks += 4) {
    const int kb = ks + hi * 2;
    v2f a;  a.x  = pa[kb]; a.y  = pa[kb + 1];
    v2f bb; bb.x = pb[kb]; bb.y = pb[kb + 1];
    bias = __builtin_amdgcn_wmma_f32_16x16x4_f32(
        false, a, false, bb, (short)0, bias, false, false);
  }
  bias *= ps;

  const bf16* qrow = Q  + ((size_t)b * SEQ + q0 + lm) * D_MODEL;
  const bf16* krow = Kp + ((size_t)b * SEQ + k0 + lm) * D_MODEL;
#pragma unroll 1
  for (int h = 0; h < NHEAD; ++h) {
    v8f acc = bias;
#pragma unroll
    for (int kk0 = 0; kk0 < HEAD_DIM; kk0 += 32) {
      const int ka = h * HEAD_DIM + kk0 + hi * 8;
      v16bf af = cat8(*(const bf16x8*)(qrow + ka), *(const bf16x8*)(qrow + ka + 16));
      const int kb = h * HEAD_DIM + kk0 + hi * 8;
      v16bf bfg = cat8(*(const bf16x8*)(krow + kb), *(const bf16x8*)(krow + kb + 16));
      acc = __builtin_amdgcn_wmma_f32_16x16x32_bf16(
          false, af, false, bfg, (short)0, acc, false, false);
    }
    float* out = attnW + (((size_t)(b * NHEAD + h) * SEQ + q0) * SEQ) + k0;
#pragma unroll
    for (int r = 0; r < 8; ++r)
      out[(size_t)(r + hi * 8) * SEQ + lm] = acc[r];
  }
}

// ---------------- softmax in place: one wave per row of 1024 ----------------
__global__ __launch_bounds__(256) void softmax_kernel(float* __restrict__ attnW) {
  const int row  = blockIdx.x * 8 + (threadIdx.x >> 5);
  const int lane = threadIdx.x & 31;
  float* p = attnW + (size_t)row * SEQ;
  float v[32];
  float m = -INFINITY;
#pragma unroll
  for (int i = 0; i < 32; ++i) { v[i] = p[lane + i * 32]; m = fmaxf(m, v[i]); }
#pragma unroll
  for (int off = 16; off > 0; off >>= 1) m = fmaxf(m, __shfl_xor(m, off, 32));
  float s = 0.0f;
#pragma unroll
  for (int i = 0; i < 32; ++i) { v[i] = __expf(v[i] - m); s += v[i]; }
#pragma unroll
  for (int off = 16; off > 0; off >>= 1) s += __shfl_xor(s, off, 32);
  const float inv = 1.0f / s;
#pragma unroll
  for (int i = 0; i < 32; ++i) p[lane + i * 32] = v[i] * inv;
}

// ---------------- PV: AO[b, q, h*64+d] = sum_k P[b,h,q,k] * Vt[b,h,d,k] -------
__global__ __launch_bounds__(32) void pv_kernel(
    const float* __restrict__ attnW, const bf16* __restrict__ Vt,
    bf16* __restrict__ AO) {
  const int lane = threadIdx.x;
  const int lm = lane & 15;
  const int hi = lane >> 4;
  const int q0 = blockIdx.x * 16;
  const int h  = blockIdx.y;
  const int b  = blockIdx.z;

  const float* prow = attnW + (((size_t)(b * NHEAD + h) * SEQ) + q0 + lm) * SEQ;
  const bf16*  vth  = Vt + ((size_t)(b * NHEAD + h) * HEAD_DIM) * SEQ;
  v8f acc[4] = {};
#pragma unroll 1
  for (int kk0 = 0; kk0 < SEQ; kk0 += 32) {
    const int ka = kk0 + hi * 8;
    v4f c0 = *(const v4f*)(prow + ka);
    v4f c1 = *(const v4f*)(prow + ka + 4);
    v4f c2 = *(const v4f*)(prow + ka + 16);
    v4f c3 = *(const v4f*)(prow + ka + 20);
    v16bf af;
#pragma unroll
    for (int i = 0; i < 4; ++i) {
      af[i]      = (bf16)c0[i];
      af[4 + i]  = (bf16)c1[i];
      af[8 + i]  = (bf16)c2[i];
      af[12 + i] = (bf16)c3[i];
    }
    const int kb = kk0 + hi * 8;
#pragma unroll
    for (int f = 0; f < 4; ++f) {
      const bf16* vr = vth + (size_t)(f * 16 + lm) * SEQ;
      v16bf bfg = cat8(*(const bf16x8*)(vr + kb), *(const bf16x8*)(vr + kb + 16));
      acc[f] = __builtin_amdgcn_wmma_f32_16x16x32_bf16(
          false, af, false, bfg, (short)0, acc[f], false, false);
    }
  }
#pragma unroll
  for (int f = 0; f < 4; ++f)
#pragma unroll
    for (int r = 0; r < 8; ++r)
      AO[((size_t)(b * SEQ + q0 + r + hi * 8)) * D_MODEL + h * HEAD_DIM + f * 16 + lm] =
          (bf16)acc[f][r];
}

extern "C" void kernel_launch(void* const* d_in, const int* in_sizes, int n_in,
                              void* d_out, int out_size, void* d_ws, size_t ws_size,
                              hipStream_t stream) {
  (void)in_sizes; (void)n_in; (void)out_size; (void)ws_size;
  const float* content = (const float*)d_in[0];
  const float* phase   = (const float*)d_in[1];
  const float* Wq = (const float*)d_in[2];
  const float* bq = (const float*)d_in[3];
  const float* Wk = (const float*)d_in[4];
  const float* bk = (const float*)d_in[5];
  const float* Wv = (const float*)d_in[6];
  const float* bv = (const float*)d_in[7];
  const float* Wo = (const float*)d_in[8];
  const float* bo = (const float*)d_in[9];
  const float* ps = (const float*)d_in[10];

  float* outMain = (float*)d_out;                               // [4,1024,1024]
  float* attnW   = outMain + (size_t)BB * SEQ * D_MODEL;        // [4,16,1024,1024]

  char* ws = (char*)d_ws;
  size_t off = 0;
  auto alloc = [&](size_t bytes) {
    char* p = ws + off;
    off = (off + bytes + 255) & ~(size_t)255;
    return p;
  };
  const size_t actB = (size_t)MTOT * D_MODEL * sizeof(bf16);    // 8 MB
  const size_t wB   = (size_t)D_MODEL * D_MODEL * sizeof(bf16); // 2 MB
  bf16* contentB = (bf16*)alloc(actB);
  bf16* WqB = (bf16*)alloc(wB);
  bf16* WkB = (bf16*)alloc(wB);
  bf16* WvB = (bf16*)alloc(wB);
  bf16* WoB = (bf16*)alloc(wB);
  bf16* Qb  = (bf16*)alloc(actB);
  bf16* Kb  = (bf16*)alloc(actB);
  bf16* Vb  = (bf16*)alloc(actB);
  bf16* Vt  = (bf16*)alloc(actB);
  bf16* AOb = (bf16*)alloc(actB);
  float* pn = (float*)alloc((size_t)BB * SEQ * PHASE_DIM * sizeof(float));

  const int nAct = MTOT * D_MODEL;       // 4194304
  const int nW   = D_MODEL * D_MODEL;    // 1048576
  cvt_f32_bf16_kernel<<<(nAct + 255) / 256, 256, 0, stream>>>(content, contentB, nAct);
  cvt_f32_bf16_kernel<<<(nW + 255) / 256, 256, 0, stream>>>(Wq, WqB, nW);
  cvt_f32_bf16_kernel<<<(nW + 255) / 256, 256, 0, stream>>>(Wk, WkB, nW);
  cvt_f32_bf16_kernel<<<(nW + 255) / 256, 256, 0, stream>>>(Wv, WvB, nW);
  cvt_f32_bf16_kernel<<<(nW + 255) / 256, 256, 0, stream>>>(Wo, WoB, nW);

  phase_norm_kernel<<<BB * SEQ, 32, 0, stream>>>(phase, pn);

  dim3 gg(D_MODEL / 64, MTOT / 64);      // (16, 64), 128 threads = 4 waves
  gemm_nt_tdm_kernel<true><<<gg, 128, 0, stream>>>(contentB, WqB, bq, Qb, 0.125f); // 1/sqrt(64)
  gemm_nt_tdm_kernel<true><<<gg, 128, 0, stream>>>(contentB, WkB, bk, Kb, 1.0f);
  gemm_nt_tdm_kernel<true><<<gg, 128, 0, stream>>>(contentB, WvB, bv, Vb, 1.0f);

  transpose_v_kernel<<<(nAct + 255) / 256, 256, 0, stream>>>(Vb, Vt);

  dim3 gl(SEQ / 16, SEQ / 16, BB);       // (64, 64, 4)
  logits_kernel<<<gl, 32, 0, stream>>>(Qb, Kb, pn, ps, attnW);

  softmax_kernel<<<(BB * NHEAD * SEQ) / 8, 256, 0, stream>>>(attnW);

  dim3 gp(SEQ / 16, NHEAD, BB);          // (64, 16, 4)
  pv_kernel<<<gp, 32, 0, stream>>>(attnW, Vt, AOb);

  gemm_nt_tdm_kernel<false><<<gg, 128, 0, stream>>>(AOb, WoB, bo, outMain, 1.0f);
}
